// RNN_43301860278765
// MI455X (gfx1250) — matrix operand order repaired and checked by reading, not compiled, not run on hardware
//
#include <hip/hip_runtime.h>
#include <cstdint>
#include <cstddef>

// Problem sizes (match reference)
#define TT    256
#define BB    32
#define VOCABN 32000
#define EMBN  256
#define LATN  512
#define OUTN  32000
#define M1    (TT * BB)        // 8192 rows of (t,b)

typedef __attribute__((ext_vector_type(2)))  float    v2f;
typedef __attribute__((ext_vector_type(8)))  float    v8f;
typedef __attribute__((ext_vector_type(8)))  __bf16   v8bf;
typedef __attribute__((ext_vector_type(16))) __bf16   v16bf;
typedef __attribute__((ext_vector_type(4)))  unsigned tdm_u4;
typedef __attribute__((ext_vector_type(8)))  int      tdm_i8;
typedef __attribute__((ext_vector_type(4)))  int      tdm_i4;

#define HAVE_TDM __has_builtin(__builtin_amdgcn_tensor_load_to_lds)

__device__ __forceinline__ uint16_t f2bf_bits(float f) {
  union { float f; uint32_t u; } v; v.f = f;
  uint32_t u = v.u;
  uint32_t r = u + 0x7FFFu + ((u >> 16) & 1u);   // round-to-nearest-even
  return (uint16_t)(r >> 16);
}

// Build v16bf from two 16B chunks (A and B operands: +16 element gap)
__device__ __forceinline__ v16bf pack16(const uint16_t* lo_p, const uint16_t* hi_p) {
  v8bf lo = *(const v8bf*)lo_p;
  v8bf hi = *(const v8bf*)hi_p;
  v16bf r;
#pragma unroll
  for (int j = 0; j < 8; ++j) { r[j] = lo[j]; r[j + 8] = hi[j]; }
  return r;
}

// ---- CDNA5 LDS addressing + async copy (ASYNCcnt) -------------------------
__device__ __forceinline__ unsigned lds_off(const void* p) {
  // generic -> LDS address space; AS3 pointers are byte offsets from LDS base
  return (unsigned)(uintptr_t)(const __attribute__((address_space(3))) void*)p;
}

__device__ __forceinline__ void async_ld_b128(unsigned lds_byte_off, const void* gaddr) {
  asm volatile("global_load_async_to_lds_b128 %0, %1, off"
               :: "v"(lds_byte_off), "v"((uint64_t)(uintptr_t)gaddr)
               : "memory");
}

__device__ __forceinline__ void wait_async0() {
#if __has_builtin(__builtin_amdgcn_s_wait_asynccnt)
  __builtin_amdgcn_s_wait_asynccnt(0);
#else
  asm volatile("s_wait_asynccnt 0" ::: "memory");
#endif
}

__device__ __forceinline__ void wait_tensor0() {
#if __has_builtin(__builtin_amdgcn_s_wait_tensorcnt)
  __builtin_amdgcn_s_wait_tensorcnt(0);
#else
  asm volatile("s_wait_tensorcnt 0" ::: "memory");
#endif
}

// ---- Tensor Data Mover: 2D tile (bf16) global -> LDS ----------------------
// tile_d0 elems per row (must give 16-DWORD rows for the pad config below),
// tile_d1 rows, row stride `stride` elems.  LDS rows padded by 4 DWORDs.
#if HAVE_TDM
__device__ __forceinline__ void tdm_load_tile_bf16(unsigned lds_byte_off,
                                                   const uint16_t* gaddr,
                                                   unsigned tile_d0, unsigned tile_d1,
                                                   unsigned stride) {
  uint64_t ga = (uint64_t)(uintptr_t)gaddr;
  tdm_u4 g0;
  g0[0] = 1u;                                            // count=1, user D#
  g0[1] = lds_byte_off;                                  // lds_addr
  g0[2] = (unsigned)(ga & 0xFFFFFFFFu);                  // global_addr[31:0]
  g0[3] = (unsigned)((ga >> 32) & 0x1FFFFFFu)            // global_addr[56:32]
        | (2u << 30);                                    // type = 2 ("image")
  tdm_i8 g1;
  g1[0] = (int)((1u << 16)      // data_size = 2 bytes
              | (1u << 20)      // pad_enable
              | (3u << 22)      // pad_interval: 16 DWORDs (64B rows)
              | (3u << 25));    // pad_amount:   4 DWORDs (16B)
  g1[1] = (int)((stride & 0xFFFFu) << 16);               // tensor_dim0[15:0]
  g1[2] = (int)(((stride >> 16) & 0xFFFFu)               // tensor_dim0[31:16]
              | ((tile_d1 & 0xFFFFu) << 16));            // tensor_dim1[15:0]
  g1[3] = (int)(((tile_d1 >> 16) & 0xFFFFu)              // tensor_dim1[31:16]
              | ((tile_d0 & 0xFFFFu) << 16));            // tile_dim0
  g1[4] = (int)(tile_d1 & 0xFFFFu);                      // tile_dim1 (tile_dim2 = 0)
  g1[5] = (int)stride;                                   // tensor_dim0_stride[31:0]
  g1[6] = 0;                                             // stride[47:32] | dim1_stride lo
  g1[7] = 0;
  tdm_i4 gz = {0, 0, 0, 0};
#if defined(__clang_major__) && (__clang_major__ >= 23)
  tdm_i8 gz8 = {0, 0, 0, 0, 0, 0, 0, 0};
  __builtin_amdgcn_tensor_load_to_lds(g0, g1, gz, gz, gz8, 0);
#else
  __builtin_amdgcn_tensor_load_to_lds(g0, g1, gz, gz, 0);
#endif
}
#endif

// ---------------------------------------------------------------------------
// Kernel 0: zero a few u32 (barrier words)
// ---------------------------------------------------------------------------
__global__ void zero_u32_kernel(unsigned* p, int n) {
  int i = blockIdx.x * 256 + threadIdx.x;
  if (i < n) p[i] = 0u;
}

// ---------------------------------------------------------------------------
// Kernel 1: convert + transpose W_o (LAT x OUT f32) -> WoT (OUT x LAT bf16)
// 32x32 tile through LDS; tile rows fetched with async-to-LDS b128 copies.
// Grid (LAT/32 = 16, OUT/32 = 1000), block 256.
// ---------------------------------------------------------------------------
__global__ void convert_wo_kernel(const float* __restrict__ Wo,
                                  uint16_t* __restrict__ WoT) {
  __shared__ float tile[32][36];            // pitch 36 dwords (16B aligned rows)
  const int k0 = blockIdx.x * 32;
  const int n0 = blockIdx.y * 32;
  const int tid = threadIdx.x;
  const int tx = tid & 31;
  const int ty = tid >> 5;                  // 0..7

  // 32 rows x 4 x 16B segments (cols 0..15) = 128 async copies, 2 each by threads 0..63
  if (tid < 64) {
    int row = tid >> 1, seg = tid & 1;
    async_ld_b128(lds_off(&tile[row][seg * 4]),
                  Wo + (size_t)(k0 + row) * OUTN + n0 + seg * 4 + 0);
    async_ld_b128(lds_off(&tile[row][seg * 4 + 8]),
                  Wo + (size_t)(k0 + row) * OUTN + n0 + seg * 4 + 8);
  }
  wait_async0();
  __syncthreads();
#pragma unroll
  for (int r = 0; r < 32; r += 8) {
    int n = ty + r;
    WoT[(size_t)(n0 + n) * LATN + (k0 + tx)] = f2bf_bits(tile[tx][n]);
  }
}

// ---------------------------------------------------------------------------
// Kernel 2: embedding gather fused with xproj = emb[x] @ W_x + b_h
// f32 WMMA 16x16x4.  Grid: (8, 512), block 128 (4 waves, 1 nTile each).
// ---------------------------------------------------------------------------
__global__ void embed_xproj_kernel(const int*   __restrict__ x,
                                   const float* __restrict__ emb,
                                   const float* __restrict__ Wx,
                                   const float* __restrict__ bh,
                                   float*       __restrict__ xproj)
{
  const int lane  = threadIdx.x & 31;
  const int wave  = threadIdx.x >> 5;
  const int nTile = blockIdx.x * 4 + wave;   // 0..31
  const int mTile = blockIdx.y;              // 0..511
  const int half  = lane >> 4;
  const int l15   = lane & 15;
  const int col   = nTile * 16 + l15;
  const int m     = mTile * 16 + l15;

  const int tok = x[m];
  const float* arow = emb + (size_t)tok * EMBN;

  v8f acc;
  const float bias = bh[col];
#pragma unroll
  for (int j = 0; j < 8; ++j) acc[j] = bias;

#pragma unroll 4
  for (int k = 0; k < EMBN; k += 4) {
    v2f a = *(const v2f*)(arow + k + 2 * half);
    v2f b;
    b.x = Wx[(size_t)(k + 2 * half)     * LATN + col];
    b.y = Wx[(size_t)(k + 2 * half + 1) * LATN + col];
    acc = __builtin_amdgcn_wmma_f32_16x16x4_f32(false, a, false, b,
                                                (short)0, acc, false, false);
  }
#pragma unroll
  for (int j = 0; j < 8; ++j) {
    int row = mTile * 16 + j + 8 * half;
    xproj[(size_t)row * LATN + col] = acc[j];
  }
}

// ---------------------------------------------------------------------------
// Global barrier (sense-reversing) for the persistent scan kernel.
// ---------------------------------------------------------------------------
__device__ __forceinline__ void grid_barrier(unsigned* bar, int nblocks) {
  __threadfence();
  if (threadIdx.x == 0) {
    unsigned ph  = __atomic_load_n(bar + 1, __ATOMIC_RELAXED);
    unsigned old = atomicAdd(bar, 1u);
    if (old == (unsigned)nblocks - 1) {
      __atomic_store_n(bar, 0u, __ATOMIC_RELAXED);
      __atomic_store_n(bar + 1, ph + 1u, __ATOMIC_RELEASE);
    } else {
      while (__atomic_load_n(bar + 1, __ATOMIC_ACQUIRE) == ph) {
        __builtin_amdgcn_s_sleep(1);
      }
    }
  }
  __threadfence();
}

// ---------------------------------------------------------------------------
// Kernel 3: persistent recurrent scan (f32 WMMA 16x16x4, exact f32).
// 64 blocks x 32 threads; block = one 16x16 tile of the (32 x 512) h_t.
// W_hh column slice (512x16 f32 = 32KB) cached in LDS once.
// ---------------------------------------------------------------------------
__global__ void rnn_scan_kernel(const float* __restrict__ xproj,
                                const float* __restrict__ Whh,
                                float*       __restrict__ hid,
                                uint16_t*    __restrict__ hbf,
                                unsigned*    __restrict__ bar)
{
  __shared__ float lds_b[LATN * 16];

  const int lane  = threadIdx.x & 31;
  const int nTile = blockIdx.x & 31;
  const int mTile = blockIdx.x >> 5;
  const int half  = lane >> 4;
  const int l15   = lane & 15;
  const int col   = nTile * 16 + l15;
  const int m     = mTile * 16 + l15;

  for (int idx = lane; idx < LATN * 16; idx += 32) {
    int k = idx >> 4;
    int c = idx & 15;
    lds_b[idx] = Whh[(size_t)k * LATN + nTile * 16 + c];
  }
  __syncthreads();

  for (int t = 0; t < TT; ++t) {
    const float* xp = xproj + (size_t)t * BB * LATN;
    v8f acc;
#pragma unroll
    for (int j = 0; j < 8; ++j)
      acc[j] = xp[(size_t)(mTile * 16 + j + 8 * half) * LATN + col];

    if (t > 0) {
      const float* hp = hid + (size_t)(t - 1) * BB * LATN + (size_t)m * LATN;
#pragma unroll 4
      for (int k = 0; k < LATN; k += 4) {
        v2f a = *(const v2f*)(hp + k + 2 * half);
        v2f b;
        b.x = lds_b[(k + 2 * half)     * 16 + l15];
        b.y = lds_b[(k + 2 * half + 1) * 16 + l15];
        acc = __builtin_amdgcn_wmma_f32_16x16x4_f32(false, a, false, b,
                                                    (short)0, acc, false, false);
      }
    }

    float*    ho = hid + (size_t)t * BB * LATN;
    uint16_t* hb = hbf + (size_t)t * BB * LATN;
#pragma unroll
    for (int j = 0; j < 8; ++j) {
      int row = mTile * 16 + j + 8 * half;
      float v = acc[j] > 0.f ? acc[j] : 0.f;
      ho[(size_t)row * LATN + col] = v;
      hb[(size_t)row * LATN + col] = f2bf_bits(v);
    }

    grid_barrier(bar, 64);
  }
}

// ---------------------------------------------------------------------------
// Kernel 4: out = H(bf16) @ W_oT(bf16) + b_o  (f32 accumulate)
// Block = 256 threads (8 waves), block tile 128(M) x 128(N).
// K chunked by 32, double-buffered in LDS.  Tiles staged by the Tensor Data
// Mover (one tensor_load_to_lds per 128x32 tile, HW row padding -> PITCH=40),
// falling back to per-lane async-to-LDS copies if the builtin is missing.
// Wave (mW = w&3, nW = w>>2) computes 32x64: 2x4 grid of 16x16 WMMA tiles.
// Grid (OUT/128 = 250, M1/128 = 64).
// ---------------------------------------------------------------------------
#define KCH   32
#define PITCH 40

__global__ void out_gemm_kernel(const uint16_t* __restrict__ hbf,  // (M1, LAT)
                                const uint16_t* __restrict__ WoT,  // (OUT, LAT)
                                const float*    __restrict__ bo,
                                float*          __restrict__ out)  // (M1, OUT)
{
  __shared__ uint16_t As[2][128 * PITCH];
  __shared__ uint16_t Bs[2][128 * PITCH];

  const int tid  = threadIdx.x;
  const int lane = tid & 31;
  const int wave = tid >> 5;                 // 0..7
  const int mW   = wave & 3;                 // 0..3 : rows mW*32..+32
  const int nW   = wave >> 2;                // 0..1 : cols nW*64..+64
  const int half = lane >> 4;
  const int l15  = lane & 15;
  const int selA = half * 8;
  const int selB = half * 8;

  const int mBase = blockIdx.y * 128;
  const int nBase = blockIdx.x * 128;

  const uint16_t* gA = hbf + (size_t)mBase * LATN;   // row stride LATN
  const uint16_t* gB = WoT + (size_t)nBase * LATN;

  v8f acc[2][4];
#pragma unroll
  for (int j = 0; j < 4; ++j) {
    float bias = bo[nBase + nW * 64 + j * 16 + l15];
#pragma unroll
    for (int i = 0; i < 2; ++i)
#pragma unroll
      for (int r = 0; r < 8; ++r) acc[i][j][r] = bias;
  }

  // stage one K-chunk (A: 128x32, B: 128x32 bf16 tiles)
  auto stage = [&](int c, int buf) {
#if HAVE_TDM
    if (wave == 0) {
      tdm_load_tile_bf16(lds_off(As[buf]), gA + c * KCH, KCH, 128, LATN);
    } else if (wave == 1) {
      tdm_load_tile_bf16(lds_off(Bs[buf]), gB + c * KCH, KCH, 128, LATN);
    }
#else
#pragma unroll
    for (int r = 0; r < 2; ++r) {
      int idx = tid + 256 * r;               // 0..511
      int row = idx >> 2, seg = idx & 3;
      async_ld_b128(lds_off(As[buf] + row * PITCH + seg * 8),
                    gA + (size_t)(c * KCH) + (size_t)row * LATN + seg * 8);
      async_ld_b128(lds_off(Bs[buf] + row * PITCH + seg * 8),
                    gB + (size_t)(c * KCH) + (size_t)row * LATN + seg * 8);
    }
#endif
  };
  auto wait_stage = [&]() {
#if HAVE_TDM
    wait_tensor0();
#else
    wait_async0();
#endif
  };

  stage(0, 0);
  wait_stage();
  __syncthreads();

  const int nChunks = LATN / KCH;            // 16
  for (int c = 0; c < nChunks; ++c) {
    int cur = c & 1;
    if (c + 1 < nChunks) stage(c + 1, cur ^ 1);

    const uint16_t* A = As[cur];
    const uint16_t* B = Bs[cur];
    v16bf a[2], b[4];
#pragma unroll
    for (int i = 0; i < 2; ++i) {
      const uint16_t* p = A + (mW * 32 + i * 16 + l15) * PITCH + selA;
      a[i] = pack16(p, p + 16);
    }
#pragma unroll
    for (int j = 0; j < 4; ++j) {
      const uint16_t* p = B + (nW * 64 + j * 16 + l15) * PITCH + selB;
      b[j] = pack16(p, p + 16);
    }
#pragma unroll
    for (int i = 0; i < 2; ++i)
#pragma unroll
      for (int j = 0; j < 4; ++j)
        acc[i][j] = __builtin_amdgcn_wmma_f32_16x16x32_bf16(
            false, a[i], false, b[j], (short)0, acc[i][j], false, false);

    wait_stage();
    __syncthreads();
  }

#pragma unroll
  for (int i = 0; i < 2; ++i) {
#pragma unroll
    for (int j = 0; j < 4; ++j) {
      int colj = nBase + nW * 64 + j * 16 + l15;
#pragma unroll
      for (int r = 0; r < 8; ++r) {
        int row = mBase + mW * 32 + i * 16 + r + 8 * half;
        out[(size_t)row * OUTN + colj] = acc[i][j][r];
      }
    }
  }
}

// ---------------------------------------------------------------------------
// Host launcher
// ---------------------------------------------------------------------------
extern "C" void kernel_launch(void* const* d_in, const int* in_sizes, int n_in,
                              void* d_out, int out_size, void* d_ws, size_t ws_size,
                              hipStream_t stream) {
  (void)in_sizes; (void)n_in; (void)out_size; (void)ws_size;

  const int*   x   = (const int*)d_in[0];
  const float* emb = (const float*)d_in[1];
  const float* W_h = (const float*)d_in[2];
  const float* b_h = (const float*)d_in[3];
  const float* W_o = (const float*)d_in[4];
  const float* b_o = (const float*)d_in[5];

  float* out = (float*)d_out;                       // (M1, OUT)
  float* hid = out + (size_t)M1 * OUTN;             // (M1, LAT)

  char* ws = (char*)d_ws;
  float*    xproj = (float*)ws;                                             // 16 MB
  uint16_t* hbf   = (uint16_t*)(ws + (size_t)M1 * LATN * 4);                //  8 MB
  uint16_t* WoT   = (uint16_t*)(ws + (size_t)M1 * LATN * 4
                                   + (size_t)M1 * LATN * 2);                // 32 MB
  unsigned* bar   = (unsigned*)(ws + (size_t)M1 * LATN * 4
                                   + (size_t)M1 * LATN * 2
                                   + (size_t)OUTN * LATN * 2);              //  8 B

  const float* W_x  = W_h;
  const float* W_hh = W_h + (size_t)EMBN * LATN;

  zero_u32_kernel<<<1, 256, 0, stream>>>(bar, 2);

  convert_wo_kernel<<<dim3(LATN / 32, OUTN / 32), 256, 0, stream>>>(W_o, WoT);

  embed_xproj_kernel<<<dim3(8, M1 / 16), 128, 0, stream>>>(x, emb, W_x, b_h, xproj);

  rnn_scan_kernel<<<64, 32, 0, stream>>>(xproj, W_hh, hid, hbf, bar);

  out_gemm_kernel<<<dim3(OUTN / 128, M1 / 128), 256, 0, stream>>>(hbf, WoT, b_o, out);
}
